// VerilogCompatibleBNN_34617436406364
// MI455X (gfx1250) — 8-bit matrix operand order repaired and checked by reading, compiled, not run on hardware
//
#include <hip/hip_runtime.h>
#include <cstdint>
#include <cstddef>


typedef __attribute__((ext_vector_type(8))) int v8i;

constexpr int BSZ = 2048;
constexpr int IN  = 4096;
constexpr int HID = 4096;
constexpr int OUT = 1024;
constexpr int TPB = 256;

__global__ __launch_bounds__(TPB)
void binarize_rowsum(const float* __restrict__ x, uint8_t* __restrict__ xb,
                     int* __restrict__ rowsum, int K) {
  __shared__ int sdata[TPB];
  const int row = blockIdx.x;
  const float*  xr  = x  + (size_t)row * K;
  uint8_t*      xbr = xb + (size_t)row * K;
  int local = 0;
  for (int i = threadIdx.x; i < K; i += TPB) {
    const int b = (xr[i] >= 0.0f) ? 1 : 0;
    xbr[i] = (uint8_t)b;
    local += b;
  }
  sdata[threadIdx.x] = local;
  __syncthreads();
  for (int s = TPB / 2; s > 0; s >>= 1) {
    if (threadIdx.x < s) sdata[threadIdx.x] += sdata[threadIdx.x + s];
    __syncthreads();
  }
  if (threadIdx.x == 0) rowsum[row] = sdata[0];
}

__global__ __launch_bounds__(TPB)
void pack_rowsum_i32(const int* __restrict__ w, uint8_t* __restrict__ wu8,
                     int* __restrict__ rowsum, int K) {
  __shared__ int sdata[TPB];
  const int row = blockIdx.x;
  const int* wr  = w   + (size_t)row * K;
  uint8_t*   wu  = wu8 + (size_t)row * K;
  int local = 0;
  for (int i = threadIdx.x; i < K; i += TPB) {
    const int b = wr[i] & 1;
    wu[i] = (uint8_t)b;
    local += b;
  }
  sdata[threadIdx.x] = local;
  __syncthreads();
  for (int s = TPB / 2; s > 0; s >>= 1) {
    if (threadIdx.x < s) sdata[threadIdx.x] += sdata[threadIdx.x + s];
    __syncthreads();
  }
  if (threadIdx.x == 0) rowsum[row] = sdata[0];
}

__global__ __launch_bounds__(TPB)
void rowsum_u8(const uint8_t* __restrict__ a, int* __restrict__ rowsum, int K) {
  __shared__ int sdata[TPB];
  const int row = blockIdx.x;
  const uint32_t* ar = (const uint32_t*)(a + (size_t)row * K);
  int local = 0;
  for (int i = threadIdx.x; i < (K >> 2); i += TPB) {
    const uint32_t v = ar[i];
    local += (int)((v & 0xFFu) + ((v >> 8) & 0xFFu) + ((v >> 16) & 0xFFu) + (v >> 24));
  }
  sdata[threadIdx.x] = local;
  __syncthreads();
  for (int s = TPB / 2; s > 0; s >>= 1) {
    if (threadIdx.x < s) sdata[threadIdx.x] += sdata[threadIdx.x + s];
    __syncthreads();
  }
  if (threadIdx.x == 0) rowsum[row] = sdata[0];
}

template <bool WRITE_ACT>
__global__ __launch_bounds__(TPB)
void bnn_gemm_wmma(const uint8_t* __restrict__ A,
                   const uint8_t* __restrict__ Wt,
                   const int* __restrict__ sumA,
                   const int* __restrict__ sumW,
                   const int* __restrict__ thrP,
                   uint8_t* __restrict__ actOut,
                   float* __restrict__ fOut,
                   int N, int K) {
  const int lane   = threadIdx.x & 31;
  const int wave   = threadIdx.x >> 5;
  const int wave_m = wave & 3;
  const int wave_n = wave >> 2;
  const int m_base = blockIdx.y * 128 + wave_m * 32;
  const int n_base = blockIdx.x * 64  + wave_n * 32;

  const int half = lane >> 4;
  const int l16  = lane & 15;

  const uint8_t* aRow0 = A  + (size_t)(m_base + l16) * K;
  const uint8_t* aRow1 = aRow0 + (size_t)16 * K;
  const uint8_t* bRow0 = Wt + (size_t)(n_base + l16) * K;
  const uint8_t* bRow1 = bRow0 + (size_t)16 * K;

  v8i acc[2][2] = {};

  for (int k0 = 0; k0 < K; k0 += 64) {
    v8i afrag[2], bfrag[2];
#pragma unroll
    for (int v = 0; v < 8; ++v) {
      const int aoff = k0 + ((v >> 1) << 4) + ((v & 1) << 2) + (half << 3);
      afrag[0][v] = *(const int*)(aRow0 + aoff);
      afrag[1][v] = *(const int*)(aRow1 + aoff);
      const int boff = k0 + ((v >> 1) << 4) + ((v & 1) << 2) + (half << 3);
      bfrag[0][v] = *(const int*)(bRow0 + boff);
      bfrag[1][v] = *(const int*)(bRow1 + boff);
    }
#pragma unroll
    for (int mt = 0; mt < 2; ++mt)
#pragma unroll
      for (int nt = 0; nt < 2; ++nt)
        acc[mt][nt] = __builtin_amdgcn_wmma_i32_16x16x64_iu8(
             false, afrag[mt],  false, bfrag[nt],
            acc[mt][nt],  false,  false);
    if (k0 + 64 < K) {
      __builtin_prefetch(aRow0 + k0 + 64, 0, 3);
      __builtin_prefetch(aRow1 + k0 + 64, 0, 3);
      __builtin_prefetch(bRow0 + k0 + 64, 0, 3);
      __builtin_prefetch(bRow1 + k0 + 64, 0, 3);
    }
  }

  const int thr = WRITE_ACT ? *thrP : 0;
#pragma unroll
  for (int mt = 0; mt < 2; ++mt) {
#pragma unroll
    for (int nt = 0; nt < 2; ++nt) {
      const int n  = n_base + nt * 16 + l16;
      const int sw = sumW[n];
#pragma unroll
      for (int r = 0; r < 8; ++r) {
        const int m  = m_base + mt * 16 + half * 8 + r;
        const int pc = 2 * acc[mt][nt][r] - sumA[m] - sw + K;
        if (WRITE_ACT) {
          actOut[(size_t)m * N + n] = (uint8_t)(pc >= thr);
        } else {
          fOut[(size_t)m * N + n] = (float)pc;
        }
      }
    }
  }
}

extern "C" void kernel_launch(void* const* d_in, const int* in_sizes, int n_in,
                              void* d_out, int out_size, void* d_ws, size_t ws_size,
                              hipStream_t stream) {
  (void)in_sizes; (void)n_in; (void)out_size; (void)ws_size;
  const float* x   = (const float*)d_in[0];
  const int*   fc1 = (const int*)d_in[1];
  const int*   fc2 = (const int*)d_in[2];
  const int*   thr = (const int*)d_in[3];
  float*       out = (float*)d_out;

  uint8_t* p    = (uint8_t*)d_ws;
  uint8_t* xb   = p;                 p += (size_t)BSZ * IN;
  uint8_t* w1   = p;                 p += (size_t)HID * IN;
  uint8_t* w2   = p;                 p += (size_t)OUT * HID;
  uint8_t* act1 = p;                 p += (size_t)BSZ * HID;
  int* Sa   = (int*)p;               p += sizeof(int) * BSZ;
  int* Sw1  = (int*)p;               p += sizeof(int) * HID;
  int* Sw2  = (int*)p;               p += sizeof(int) * OUT;
  int* Sact = (int*)p;

  binarize_rowsum<<<BSZ, TPB, 0, stream>>>(x, xb, Sa, IN);
  pack_rowsum_i32<<<HID, TPB, 0, stream>>>(fc1, w1, Sw1, IN);
  pack_rowsum_i32<<<OUT, TPB, 0, stream>>>(fc2, w2, Sw2, HID);

  bnn_gemm_wmma<true><<<dim3(HID / 64, BSZ / 128), TPB, 0, stream>>>(
      xb, w1, Sa, Sw1, thr, act1, nullptr, HID, IN);

  rowsum_u8<<<BSZ, TPB, 0, stream>>>(act1, Sact, HID);

  bnn_gemm_wmma<false><<<dim3(OUT / 64, BSZ / 128), TPB, 0, stream>>>(
      act1, w2, Sact, Sw2, nullptr, nullptr, out, OUT, HID);
}
